// Attention_22497038697095
// MI455X (gfx1250) — matrix operand order repaired and checked by reading, not compiled, not run on hardware
//
#include <hip/hip_runtime.h>

// ---------------------------------------------------------------------------
// CDNA5 (gfx1250) attention block.
//  - one-time fp32->bf16 convert (+weight transpose) prepass
//  - bf16 WMMA GEMMs and flash-style attention (v_wmma_f32_16x16x32_bf16)
//  - Tensor Data Mover (tensor_load_to_lds) double-buffered LDS staging,
//    with D# pad fields generating the padded LDS row strides in hardware
// ---------------------------------------------------------------------------

typedef __bf16 bf16;
typedef __attribute__((ext_vector_type(16))) __bf16 bf16x16;
typedef __attribute__((ext_vector_type(8)))  __bf16 bf16x8;
typedef __attribute__((ext_vector_type(8)))  float  f32x8;
typedef __attribute__((ext_vector_type(4)))  unsigned uint32x4;
typedef __attribute__((ext_vector_type(8)))  int      int32x8;
typedef __attribute__((ext_vector_type(4)))  int      int32x4;

#if __has_builtin(__builtin_amdgcn_tensor_load_to_lds) && \
    __has_builtin(__builtin_amdgcn_s_wait_tensorcnt)
#define USE_TDM 1
#else
#define USE_TDM 0
#endif

union Frag { bf16x16 v; bf16x8 h[2]; };

__device__ inline unsigned f2bfbits(float f) {
  unsigned u = __builtin_bit_cast(unsigned, f);
  u += 0x7FFFu + ((u >> 16) & 1u);          // round-to-nearest-even
  return u >> 16;
}
__device__ inline bf16 f2bf(float f) {
  unsigned short s = (unsigned short)f2bfbits(f);
  return __builtin_bit_cast(bf16, s);
}

__device__ inline bf16x8 ld16b(const bf16* p) {
  return *(const bf16x8*)__builtin_assume_aligned(p, 16);
}

__device__ inline f32x8 wmma_bf16(bf16x16 a, bf16x16 b, f32x8 c) {
  return __builtin_amdgcn_wmma_f32_16x16x32_bf16(false, a, false, b,
                                                 (short)0, c, false, false);
}

__device__ inline float rmax16(float v) {
  v = fmaxf(v, __shfl_xor(v, 1, 16));
  v = fmaxf(v, __shfl_xor(v, 2, 16));
  v = fmaxf(v, __shfl_xor(v, 4, 16));
  v = fmaxf(v, __shfl_xor(v, 8, 16));
  return v;
}
__device__ inline float rsum16(float v) {
  v += __shfl_xor(v, 1, 16);
  v += __shfl_xor(v, 2, 16);
  v += __shfl_xor(v, 4, 16);
  v += __shfl_xor(v, 8, 16);
  return v;
}

// ---------------------------------------------------------------------------
// TDM: 2D bf16 tile (tile0 x tile1) from global -> LDS, with LDS row padding.
// D# layout per CDNA5 ISA ch.8: group0 {count,lds_addr,global_addr,type=2},
// group1 {data_size=2B, pad_enable, pad codes, dims, tile dims, stride}.
// lds_byte is the byte offset inside this kernel's (single) LDS block.
// clang-23 toolchain builtin form: 6 args (g0, g1, g2, g3, g4, cpol).
// ---------------------------------------------------------------------------
__device__ inline void tdm_load_2d(unsigned lds_byte, const void* gptr,
                                   unsigned td0, unsigned td1,
                                   unsigned tile0, unsigned tile1,
                                   unsigned stride0,
                                   unsigned padi, unsigned pada) {
#if USE_TDM
  unsigned long long ga = (unsigned long long)gptr;
  uint32x4 g0 = { 1u,                                  // count=1 (user D#)
                  lds_byte,
                  (unsigned)ga,
                  (unsigned)((ga >> 32) & 0x01FFFFFFull) | (2u << 30) };
  unsigned w0 = (1u << 16)            // data_size = 1 -> 2 bytes
              | (1u << 20)            // pad_enable
              | (padi << 22) | (pada << 25);
  int32x8 g1 = { (int)w0,
                 (int)(td0 << 16),                       // tensor_dim0 lo
                 (int)((td0 >> 16) | (td1 << 16)),       // td0 hi | td1 lo
                 (int)((td1 >> 16) | (tile0 << 16)),     // td1 hi | tile_dim0
                 (int)tile1,                             // tile_dim1 (dim2=0)
                 (int)stride0,                           // dim0 stride lo
                 0, 0 };
  int32x4 z4 = {0, 0, 0, 0};
  int32x8 z8i = {0, 0, 0, 0, 0, 0, 0, 0};
  __builtin_amdgcn_tensor_load_to_lds(g0, g1, z4, z4, z8i, 0);
#else
  (void)lds_byte; (void)gptr; (void)td0; (void)td1;
  (void)tile0; (void)tile1; (void)stride0; (void)padi; (void)pada;
#endif
}

// ---------------------------------------------------------------------------
// Prepass kernels: fp32 -> bf16 copy, and fp32 [K][N] -> bf16 [N][K] transpose
// ---------------------------------------------------------------------------
__global__ __launch_bounds__(256) void cvt_copy_kernel(
    const float* __restrict__ in, void* __restrict__ outv, int n) {
  bf16* out = (bf16*)outv;
  int i = (blockIdx.x * 256 + threadIdx.x) * 4;
  if (i < n) {
    float4 v = *(const float4*)(in + i);
    uint2 o;
    o.x = (f2bfbits(v.y) << 16) | f2bfbits(v.x);
    o.y = (f2bfbits(v.w) << 16) | f2bfbits(v.z);
    *(uint2*)(out + i) = o;
  }
}

__global__ __launch_bounds__(256) void cvt_transpose_kernel(
    const float* __restrict__ in, void* __restrict__ outv, int K, int N) {
  bf16* out = (bf16*)outv;
  __shared__ bf16 T[64][65];
  const int tid = threadIdx.x;
  const int ntiles = N >> 6;
  const int k0 = (blockIdx.x / ntiles) << 6;
  const int n0 = (blockIdx.x % ntiles) << 6;
#pragma unroll
  for (int i = 0; i < 16; ++i) {
    int idx = tid + (i << 8);
    int r = idx >> 6, c = idx & 63;
    T[r][c] = f2bf(in[(long)(k0 + r) * N + n0 + c]);
  }
  __syncthreads();
#pragma unroll
  for (int i = 0; i < 16; ++i) {
    int idx = tid + (i << 8);
    int r = idx >> 6, c = idx & 63;
    out[(long)(n0 + r) * K + k0 + c] = T[c][r];
  }
}

// ---------------------------------------------------------------------------
// bf16 GEMM: C[M,N] = A[M,K] @ Bt[N,K]^T + bias. 64x64 block tile, BLOCK_K=32,
// 4 waves each computing 32x32 via four 16x16x32 bf16 WMMAs per K step.
// TDM double-buffered staging. MODE 0: scatter q/k/v heads (v transposed);
// MODE 1: fp32 store.
// ---------------------------------------------------------------------------
template<int MODE>
__global__ __launch_bounds__(128) void gemm_bf16_kernel(
    const void* __restrict__ Av, const void* __restrict__ Btv,
    const float* __restrict__ bias,
    void* __restrict__ qv, void* __restrict__ kv, void* __restrict__ vv,
    float* __restrict__ out, int M, int N, int K)
{
  const bf16* A  = (const bf16*)Av;
  const bf16* Bt = (const bf16*)Btv;

  struct SM { bf16 t[2][2][64][40]; };   // [buf][A/B][row][k], 80B row stride
  __shared__ __attribute__((aligned(16))) SM sm;

  const int tid  = threadIdx.x;
  const int wave = tid >> 5, lane = tid & 31;
  const int hi2  = lane >> 4, l16 = lane & 15;
  const int wm   = wave >> 1, wn = wave & 1;

  const int  ntiles = N >> 6;
  const long m0 = (long)(blockIdx.x / ntiles) << 6;
  const long n0 = (long)(blockIdx.x % ntiles) << 6;
  const int  nk = K >> 5;

  const f32x8 z8 = {0,0,0,0,0,0,0,0};
  f32x8 acc[2][2];
  acc[0][0] = z8; acc[0][1] = z8; acc[1][0] = z8; acc[1][1] = z8;

#if USE_TDM
  auto issue = [&](int ki, int buf) {
    // LDS byte offsets inside sm (LDS block starts at offset 0)
    unsigned la = (unsigned)((buf * 2 + 0) * 64 * 40 * 2);
    unsigned lb = (unsigned)((buf * 2 + 1) * 64 * 40 * 2);
    // rows of 32 bf16 = 64B, pad 16B -> interval code 3 (16 DW), amount 3 (4 DW)
    tdm_load_2d(la, A  + m0 * K + ki * 32, K, (unsigned)M, 32, 64, K, 3, 3);
    tdm_load_2d(lb, Bt + n0 * K + ki * 32, K, (unsigned)N, 32, 64, K, 3, 3);
  };
  if (wave == 0) issue(0, 0);
#endif

  for (int ki = 0; ki < nk; ++ki) {
    const int buf = ki & 1;
    __syncthreads();                       // prior reads of next buf complete
#if USE_TDM
    if (wave == 0) {
      if (ki + 1 < nk) {
        issue(ki + 1, (ki + 1) & 1);       // overlap next DMA with compute
        __builtin_amdgcn_s_wait_tensorcnt(2);   // current tile's 2 ops done
      } else {
        __builtin_amdgcn_s_wait_tensorcnt(0);
      }
    }
#else
#pragma unroll
    for (int i = 0; i < 4; ++i) {          // fallback: plain bf16 copy
      int idx = tid + (i << 7);
      int row = idx >> 3, c4 = (idx & 7) << 2;
      *(uint2*)&sm.t[buf][0][row][c4] =
          *(const uint2*)(A + (m0 + row) * K + ki * 32 + c4);
      *(uint2*)&sm.t[buf][1][row][c4] =
          *(const uint2*)(Bt + (n0 + row) * K + ki * 32 + c4);
    }
#endif
    __syncthreads();

    Frag af[2], bfr[2];
#pragma unroll
    for (int mi = 0; mi < 2; ++mi) {
      const bf16* p = &sm.t[buf][0][(wm << 5) + (mi << 4) + l16][hi2 << 3];
      af[mi].h[0] = ld16b(p);
      af[mi].h[1] = ld16b(p + 16);
    }
#pragma unroll
    for (int ni = 0; ni < 2; ++ni) {
      const bf16* p = &sm.t[buf][1][(wn << 5) + (ni << 4) + l16][hi2 << 3];
      bfr[ni].h[0] = ld16b(p);
      bfr[ni].h[1] = ld16b(p + 16);
    }
#pragma unroll
    for (int mi = 0; mi < 2; ++mi)
#pragma unroll
      for (int ni = 0; ni < 2; ++ni)
        acc[mi][ni] = wmma_bf16(af[mi].v, bfr[ni].v, acc[mi][ni]);
  }

  // epilogue: C element (row = r + 8*hi2, col = l16)
#pragma unroll
  for (int mi = 0; mi < 2; ++mi) {
#pragma unroll
    for (int ni = 0; ni < 2; ++ni) {
      long rb  = m0 + (wm << 5) + (mi << 4) + (hi2 << 3);
      long col = n0 + (wn << 5) + (ni << 4) + l16;
      float bv = bias[col];
#pragma unroll
      for (int r = 0; r < 8; ++r) {
        float v = acc[mi][ni][r] + bv;
        long row = rb + r;
        if constexpr (MODE == 0) {
          int part = (int)(col >> 10);
          int nl   = (int)(col & 1023);
          int hh   = nl >> 6, dd = nl & 63;
          long brow = row >> 9, srow = row & 511;   // row = b*512 + s
          if (part == 0)        // q: [B,H,S,dh]
            ((bf16*)qv)[((brow * 16 + hh) * 512 + srow) * 64 + dd] = f2bf(v);
          else if (part == 1)   // k: [B,H,S,dh]
            ((bf16*)kv)[((brow * 16 + hh) * 512 + srow) * 64 + dd] = f2bf(v);
          else                  // v transposed: [B,H,dh,S]  (TDM-friendly)
            ((bf16*)vv)[((brow * 16 + hh) * 64 + dd) * 512 + srow] = f2bf(v);
        } else {
          out[row * (long)N + col] = v;
        }
      }
    }
  }
}

// ---------------------------------------------------------------------------
// Flash-style attention. One block per (b, h, 64-row q tile); 4 waves, each
// wave owns 16 q rows. V tiles (already transposed in global) staged to LDS
// via TDM double buffering; QK^T and PV via bf16 WMMA; fp32 online softmax.
// ---------------------------------------------------------------------------
__global__ __launch_bounds__(128) void attn_kernel(
    const void* __restrict__ qvp, const void* __restrict__ kvp,
    const void* __restrict__ vvp, const float* __restrict__ mask,
    void* __restrict__ avp)
{
  const bf16* qg  = (const bf16*)qvp;
  const bf16* kg  = (const bf16*)kvp;
  const bf16* vgT = (const bf16*)vvp;       // [B,H,dh,S]
  bf16* aB = (bf16*)avp;

  struct SM {
    bf16 Vt[2][64][72];     // [buf][dh][key], 144B row stride (128B + 16B pad)
    bf16 Pl[4][16][72];     // per-wave P round-trip
  };
  __shared__ __attribute__((aligned(16))) SM sm;

  const int tid  = threadIdx.x;
  const int wave = tid >> 5, lane = tid & 31;
  const int hi2  = lane >> 4, l16 = lane & 15;

  const int blk = blockIdx.x;               // (b*16 + h)*8 + qt
  const int qt = blk & 7, bh = blk >> 3;
  const int h = bh & 15, b = bh >> 4;

  const bf16* qbh  = qg  + (long)bh * 512 * 64;
  const bf16* kbh  = kg  + (long)bh * 512 * 64;
  const bf16* vtbh = vgT + (long)bh * 64 * 512;
  const float* mrowp = mask + (long)b * 512;

  const int q0 = qt << 6;
  const int qw = q0 + (wave << 4);

  // Q A-fragments (16x32, dh chunks 0-31 / 32-63), loaded once from global
  Frag qf[2];
  {
    const bf16* qrow = qbh + (long)(qw + l16) * 64;
#pragma unroll
    for (int kk = 0; kk < 2; ++kk) {
      qf[kk].h[0] = ld16b(qrow + 32 * kk + (hi2 << 3));
      qf[kk].h[1] = ld16b(qrow + 32 * kk + (hi2 << 3) + 16);
    }
  }

  const f32x8 z8 = {0,0,0,0,0,0,0,0};
  f32x8 O[4]; float mrow[8], lrow[8];
#pragma unroll
  for (int dt = 0; dt < 4; ++dt) O[dt] = z8;
#pragma unroll
  for (int r = 0; r < 8; ++r) { mrow[r] = -3.0e38f; lrow[r] = 0.0f; }

  const int nkt = (q0 >> 6) + 1;            // causal: tiles 0..qt

#if USE_TDM
  auto issueV = [&](int ti, int buf) {
    // 64(dh) x 64(key) tile: rows 128B + 16B pad -> codes (4, 3); Vt at LDS 0
    tdm_load_2d((unsigned)(buf * 64 * 72 * 2), vtbh + ti * 64,
                512, 64, 64, 64, 512, 4, 3);
  };
  if (wave == 0) issueV(0, 0);
#endif

  for (int ti = 0; ti < nkt; ++ti) {
    const int kb = ti << 6;
    const int buf = ti & 1;
    __syncthreads();
#if USE_TDM
    if (wave == 0) {
      if (ti + 1 < nkt) {
        issueV(ti + 1, (ti + 1) & 1);
        __builtin_amdgcn_s_wait_tensorcnt(1);
      } else {
        __builtin_amdgcn_s_wait_tensorcnt(0);
      }
    }
#else
#pragma unroll
    for (int i = 0; i < 8; ++i) {           // fallback: bf16 copy of Vt tile
      int idx = tid + (i << 7);
      int row = idx >> 4, c4 = (idx & 15) << 2;
      *(uint2*)&sm.Vt[buf][row][c4] =
          *(const uint2*)(vtbh + (long)row * 512 + kb + c4);
    }
#endif
    __syncthreads();

    // S = Q @ K^T  (K^T B-fragments directly from global, table order/lane)
    f32x8 s[4];
#pragma unroll
    for (int nt = 0; nt < 4; ++nt) {
      const bf16* kr = kbh + (long)(kb + (nt << 4) + l16) * 64;
      f32x8 a = z8;
#pragma unroll
      for (int kk = 0; kk < 2; ++kk) {
        Frag bb;
        bb.h[0] = ld16b(kr + 32 * kk + (hi2 << 3));
        bb.h[1] = ld16b(kr + 32 * kk + (hi2 << 3) + 16);
        a = wmma_bf16(qf[kk].v, bb.v, a);
      }
      s[nt] = a;
    }

    // scale + causal/pad mask (additive -1e9, like the reference)
#pragma unroll
    for (int nt = 0; nt < 4; ++nt) {
      int kj = kb + (nt << 4) + l16;
      float mk = mrowp[kj];
#pragma unroll
      for (int r = 0; r < 8; ++r) {
        int qi = qw + r + (hi2 << 3);
        bool valid = (kj <= qi) && (mk > 0.5f);
        s[nt][r] = valid ? s[nt][r] * 0.125f : -1.0e9f;
      }
    }

    // online softmax per row (rows live in 16-lane halves)
#pragma unroll
    for (int r = 0; r < 8; ++r) {
      float tm = fmaxf(fmaxf(s[0][r], s[1][r]), fmaxf(s[2][r], s[3][r]));
      tm = rmax16(tm);
      float mn = fmaxf(mrow[r], tm);
      float al = __expf(mrow[r] - mn);
      mrow[r] = mn;
      float ls = 0.0f;
#pragma unroll
      for (int nt = 0; nt < 4; ++nt) {
        float p = __expf(s[nt][r] - mn);
        s[nt][r] = p;
        ls += p;
      }
      ls = rsum16(ls);
      lrow[r] = lrow[r] * al + ls;
#pragma unroll
      for (int dt = 0; dt < 4; ++dt) O[dt][r] *= al;
      int prow = r + (hi2 << 3);
#pragma unroll
      for (int nt = 0; nt < 4; ++nt)
        sm.Pl[wave][prow][(nt << 4) + l16] = f2bf(s[nt][r]);
    }

    // O += P @ V
    Frag pf[2];
#pragma unroll
    for (int kk = 0; kk < 2; ++kk) {
      const bf16* pp = &sm.Pl[wave][l16][32 * kk + (hi2 << 3)];
      pf[kk].h[0] = ld16b(pp);
      pf[kk].h[1] = ld16b(pp + 16);
    }
#pragma unroll
    for (int dt = 0; dt < 4; ++dt) {
#pragma unroll
      for (int kk = 0; kk < 2; ++kk) {
        Frag vb;
        const bf16* vp = &sm.Vt[buf][(dt << 4) + l16][32 * kk + (hi2 << 3)];
        vb.h[0] = ld16b(vp);
        vb.h[1] = ld16b(vp + 16);
        O[dt] = wmma_bf16(pf[kk].v, vb.v, O[dt]);
      }
    }
  }

  // finalize: O /= l, merge heads into aB[b*512+s][h*64+d] (bf16)
#pragma unroll
  for (int r = 0; r < 8; ++r) {
    float inv = 1.0f / fmaxf(lrow[r], 1e-20f);
    int sg = qw + r + (hi2 << 3);
    long rowbase = ((long)b * 512 + sg) * 1024 + h * 64;
#pragma unroll
    for (int dt = 0; dt < 4; ++dt)
      aB[rowbase + (dt << 4) + l16] = f2bf(O[dt][r] * inv);
  }
}

// ---------------------------------------------------------------------------
extern "C" void kernel_launch(void* const* d_in, const int* in_sizes, int n_in,
                              void* d_out, int out_size, void* d_ws, size_t ws_size,
                              hipStream_t stream) {
  const float* x        = (const float*)d_in[0];
  const float* mask     = (const float*)d_in[1];
  const float* c_attn_w = (const float*)d_in[2];
  const float* c_attn_b = (const float*)d_in[3];
  const float* c_proj_w = (const float*)d_in[4];
  const float* c_proj_b = (const float*)d_in[5];

  // workspace (bf16): xbf 8MB | WtA 6MB | WtP 2MB | q 8MB | k 8MB | vT 8MB |
  //                   a 8MB   (total 48MB)
  char* ws = (char*)d_ws;
  void* xbf = ws;
  void* WtA = ws + 8388608;
  void* WtP = ws + 8388608 + 6291456;
  void* qg  = ws + 16777216;
  void* kg  = ws + 25165824;
  void* vgT = ws + 33554432;
  void* aB  = ws + 41943040;

  const int M = 8 * 512;   // 4096

  // prepass: convert x, transpose+convert weights (one-time, memory-bound)
  cvt_copy_kernel<<<dim3(4096), 256, 0, stream>>>(x, xbf, M * 1024);
  cvt_transpose_kernel<<<dim3(16 * 48), 256, 0, stream>>>(c_attn_w, WtA, 1024, 3072);
  cvt_transpose_kernel<<<dim3(16 * 16), 256, 0, stream>>>(c_proj_w, WtP, 1024, 1024);

  // 1) qkv = x @ c_attn_w + b -> scatter q/k (row-major) and v (transposed)
  gemm_bf16_kernel<0><<<dim3((M / 64) * (3072 / 64)), 128, 0, stream>>>(
      xbf, WtA, c_attn_b, qg, kg, vgT, nullptr, M, 3072, 1024);

  // 2) flash attention per (b, h, 64-row q tile)
  attn_kernel<<<dim3(8 * 16 * 8), 128, 0, stream>>>(qg, kg, vgT, mask, aB);

  // 3) out = a @ c_proj_w + b (fp32 output)
  gemm_bf16_kernel<1><<<dim3((M / 64) * (1024 / 64)), 128, 0, stream>>>(
      aB, WtP, c_proj_b, nullptr, nullptr, nullptr, (float*)d_out,
      M, 1024, 1024);
}
